// QWENWrapper_63402307224123
// MI455X (gfx1250) — matrix operand order repaired and checked by reading, not compiled, not run on hardware
//
#include <hip/hip_runtime.h>
#include <stdint.h>

#define SLEN    1024
#define HDIM    1024
#define NHEADS  16
#define KVHEADS 8
#define HEADDIM 128
#define FFDIM   3072
#define VOCAB   32000
#define NLAYERS 8
#define EPSV    1e-6f
#define QDIM    (NHEADS * HEADDIM)   // 2048
#define KVDIM   (KVHEADS * HEADDIM)  // 1024

typedef __attribute__((ext_vector_type(16))) __bf16 bf16x16;
typedef __attribute__((ext_vector_type(8)))  __bf16 bf16x8;
typedef __attribute__((ext_vector_type(8)))  float  f32x8;

// ---------------------------------------------------------------------------
// CDNA5 async global->LDS copy (16B per lane), tracked by ASYNCcnt.
// Flat address of a __shared__ object is {SHARED_BASE, lds_offset[31:0]},
// so the low 32 bits are the LDS byte address the instruction wants.
// ---------------------------------------------------------------------------
__device__ __forceinline__ void async_ld_b128(void* lds, const void* g) {
  unsigned loff = (unsigned)(unsigned long long)lds;
  unsigned long long ga = (unsigned long long)g;
  asm volatile("global_load_async_to_lds_b128 %0, %1, off"
               :: "v"(loff), "v"(ga) : "memory");
}
__device__ __forceinline__ void wait_async0() {
  asm volatile("s_wait_asynccnt 0x0" ::: "memory");
}

// ---------------------------------------------------------------------------
// Embedding dequant: hidden[s,h] = u8[id,h]*scale[id] + zero[id]
// ---------------------------------------------------------------------------
__global__ void embed_kernel(const int* __restrict__ ids,
                             const uint8_t* __restrict__ ed,
                             const float* __restrict__ zp,
                             const float* __restrict__ sc,
                             float* __restrict__ hidden) {
  int idx = blockIdx.x * 256 + threadIdx.x;          // S*H elements
  int s = idx >> 10, h = idx & 1023;
  int id = ids[s];
  hidden[idx] = (float)ed[(long)id * HDIM + h] * sc[id] + zp[id];
}

// ---------------------------------------------------------------------------
// RMSNorm over H=1024, one block (256 thr) per row, bf16 output
// ---------------------------------------------------------------------------
__global__ __launch_bounds__(256) void rmsnorm_bf16_kernel(
    const float* __restrict__ x, const float* __restrict__ w,
    __bf16* __restrict__ out) {
  int s = blockIdx.x;
  const float* row = x + (long)s * HDIM;
  float ss = 0.f;
  for (int h = threadIdx.x; h < HDIM; h += 256) { float v = row[h]; ss += v * v; }
  __shared__ float red[8];
  for (int o = 16; o; o >>= 1) ss += __shfl_xor(ss, o, 32);
  if ((threadIdx.x & 31) == 0) red[threadIdx.x >> 5] = ss;
  __syncthreads();
  if (threadIdx.x == 0) {
    float t = 0.f;
    for (int i = 0; i < 8; i++) t += red[i];
    red[0] = rsqrtf(t / (float)HDIM + EPSV);
  }
  __syncthreads();
  float inv = red[0];
  for (int h = threadIdx.x; h < HDIM; h += 256)
    out[(long)s * HDIM + h] = (__bf16)(row[h] * inv * w[h]);
}

// ---------------------------------------------------------------------------
// fp32 W (K x N, row major)  ->  bf16 W^T (N x K, row major)
// ---------------------------------------------------------------------------
__global__ void convert_wT_kernel(const float* __restrict__ W,
                                  __bf16* __restrict__ WT, int K, int N) {
  long idx = (long)blockIdx.x * 256 + threadIdx.x;
  if (idx >= (long)K * N) return;
  int k = (int)(idx / N), n = (int)(idx % N);
  WT[(long)n * K + k] = (__bf16)W[idx];
}

// ---------------------------------------------------------------------------
// GEMM: C[M,N](f32) = A[M,K](bf16, row major) x BT[N,K](bf16, row major)^T
// block 256 thr = 8 waves; block tile 128x128, K-slab 64; wave tile 32x64.
// LDS double-buffered; slabs staged with global_load_async_to_lds_b128.
// epilogue: 0 = store, 1 = C += acc (residual), 2 = obf = bf16(silu(gate)*acc)
// ---------------------------------------------------------------------------
__global__ __launch_bounds__(256) void gemm_bf16_wmma(
    const __bf16* __restrict__ A, const __bf16* __restrict__ BT,
    float* __restrict__ C, int M, int N, int K, int epilogue,
    const float* __restrict__ gatebuf, __bf16* __restrict__ obf) {
  __shared__ __attribute__((aligned(16))) __bf16 As[2][128 * 64];
  __shared__ __attribute__((aligned(16))) __bf16 Bs[2][128 * 64];
  int tid = threadIdx.x;
  int wid = tid >> 5, lane = tid & 31;
  int wr = wid & 3, wc = wid >> 2;          // 4 row-waves x 2 col-waves
  int g = lane >> 4, l16 = lane & 15;
  int mBase = blockIdx.y * 128, nBase = blockIdx.x * 128;

  auto stage = [&](int b, int kb) {
#pragma unroll
    for (int i = 0; i < 4; i++) {           // 1024 16B-chunks per tile, A and B each
      int cid = tid + i * 256;
      int r = cid >> 3, kc = (cid & 7) << 3;
      async_ld_b128(&As[b][r * 64 + kc], &A[(long)(mBase + r) * K + kb + kc]);
      async_ld_b128(&Bs[b][r * 64 + kc], &BT[(long)(nBase + r) * K + kb + kc]);
    }
  };

  f32x8 zero = {0.f, 0.f, 0.f, 0.f, 0.f, 0.f, 0.f, 0.f};
  f32x8 acc[2][4];
#pragma unroll
  for (int i = 0; i < 2; i++)
#pragma unroll
    for (int j = 0; j < 4; j++) acc[i][j] = zero;

  stage(0, 0);
  wait_async0();
  __syncthreads();
  int buf = 0;

  for (int kb = 0; kb < K; kb += 64) {
    if (kb + 64 < K) stage(buf ^ 1, kb + 64);   // async prefetch of next slab
#pragma unroll
    for (int ks = 0; ks < 64; ks += 32) {
      bf16x16 af[2], bfr[4];
#pragma unroll
      for (int mt = 0; mt < 2; mt++) {
        const __bf16* p = &As[buf][(wr * 32 + mt * 16 + l16) * 64 + ks + g * 16];
        bf16x8 lo = *(const bf16x8*)p;
        bf16x8 hi = *(const bf16x8*)(p + 8);
        af[mt] = __builtin_shufflevector(lo, hi, 0, 1, 2, 3, 4, 5, 6, 7, 8, 9,
                                         10, 11, 12, 13, 14, 15);
      }
#pragma unroll
      for (int nt = 0; nt < 4; nt++) {
        const __bf16* p = &Bs[buf][(wc * 64 + nt * 16 + l16) * 64 + ks + g * 16];
        bf16x8 lo = *(const bf16x8*)p;
        bf16x8 hi = *(const bf16x8*)(p + 8);
        bfr[nt] = __builtin_shufflevector(lo, hi, 0, 1, 2, 3, 4, 5, 6, 7, 8, 9,
                                          10, 11, 12, 13, 14, 15);
      }
#pragma unroll
      for (int mt = 0; mt < 2; mt++)
#pragma unroll
        for (int nt = 0; nt < 4; nt++)
          acc[mt][nt] = __builtin_amdgcn_wmma_f32_16x16x32_bf16(
              false, af[mt], false, bfr[nt], (short)0, acc[mt][nt], false, false);
    }
    wait_async0();       // next slab landed (this wave) ...
    __syncthreads();     // ... and everyone is done reading the current slab
    buf ^= 1;
  }

#pragma unroll
  for (int mt = 0; mt < 2; mt++)
#pragma unroll
    for (int nt = 0; nt < 4; nt++)
#pragma unroll
      for (int j = 0; j < 8; j++) {
        int row = mBase + wr * 32 + mt * 16 + g * 8 + j;
        int col = nBase + wc * 64 + nt * 16 + l16;
        long idx = (long)row * N + col;
        float v = acc[mt][nt][j];
        if (epilogue == 0) C[idx] = v;
        else if (epilogue == 1) C[idx] += v;
        else {
          float gv = gatebuf[idx];
          float si = gv / (1.f + __expf(-gv));
          obf[idx] = (__bf16)(si * v);
        }
      }
}

// ---------------------------------------------------------------------------
// Per-(s,head) RMSNorm over HD=128 + RoPE; src (S, nheads*HD) f32,
// dst (nheads, S, HD) bf16.  cos/sin: (MAX,HD), past_len = 0.
// ---------------------------------------------------------------------------
__global__ __launch_bounds__(128) void qk_rope_kernel(
    const float* __restrict__ src, const float* __restrict__ nw,
    const float* __restrict__ cosb, const float* __restrict__ sinb,
    __bf16* __restrict__ dst, int nheads) {
  int s = blockIdx.x / nheads;
  int h = blockIdx.x % nheads;
  int d = threadIdx.x;  // 0..127
  const float* row = src + ((long)s * nheads + h) * HEADDIM;
  float x = row[d];
  float ss = x * x;
  for (int o = 16; o; o >>= 1) ss += __shfl_xor(ss, o, 32);
  __shared__ float red[4];
  if ((d & 31) == 0) red[d >> 5] = ss;
  __syncthreads();
  if (d == 0)
    red[0] = rsqrtf((red[0] + red[1] + red[2] + red[3]) / (float)HEADDIM + EPSV);
  __syncthreads();
  float inv = red[0];
  int partner = (d < 64) ? d + 64 : d - 64;
  float xn = x * inv * nw[d];
  float xp = row[partner] * inv * nw[partner];
  float c = cosb[(long)s * HEADDIM + d];
  float sn = sinb[(long)s * HEADDIM + d];
  float rh = (d < 64) ? -xp : xp;
  dst[((long)h * SLEN + s) * HEADDIM + d] = (__bf16)(xn * c + rh * sn);
}

// v (S, KVH*HD) f32 -> vT (KVH, HD, S) bf16
__global__ void vtrans_kernel(const float* __restrict__ v, __bf16* __restrict__ vT) {
  int idx = blockIdx.x * 256 + threadIdx.x;  // KVH*S*HD total
  int s = idx / KVDIM;
  int r = idx % KVDIM;
  int h = r / HEADDIM, d = r % HEADDIM;
  vT[((long)h * HEADDIM + d) * SLEN + s] = (__bf16)v[idx];
}

// ---------------------------------------------------------------------------
// Flash attention, causal (-128 additive mask, matching reference).
// Block = 128 thr (4 waves), one (head, 64-query-tile) per block.
// Q (NH,S,HD), K (KVH,S,HD), Vt (KVH,HD,S) all bf16; out (S, NH*HD) bf16.
// K/V tiles staged with async global->LDS loads.
// ---------------------------------------------------------------------------
__global__ __launch_bounds__(128) void flash_attn_kernel(
    const __bf16* __restrict__ Q, const __bf16* __restrict__ Kc,
    const __bf16* __restrict__ Vt, __bf16* __restrict__ Obf) {
  __shared__ __attribute__((aligned(16))) __bf16 Qs[64 * 128];
  __shared__ __attribute__((aligned(16))) __bf16 Ks[64 * 128];
  __shared__ __attribute__((aligned(16))) __bf16 Vs[128 * 64];
  __shared__ __attribute__((aligned(16))) __bf16 Ps[4 * 16 * 64];
  int tid = threadIdx.x;
  int w = tid >> 5, lane = tid & 31, g = lane >> 4, l16 = lane & 15;
  const int nQT = SLEN / 64;
  int head = blockIdx.x / nQT;
  int qt = blockIdx.x % nQT;
  int kvh = head / (NHEADS / KVHEADS);
  int qBase = qt * 64;

  const __bf16* Qg = Q + ((long)head * SLEN + qBase) * HEADDIM;
#pragma unroll
  for (int i = 0; i < 8; i++) {
    int cid = tid + i * 128;
    int r = cid >> 4, kc = (cid & 15) << 3;
    async_ld_b128(&Qs[r * 128 + kc], &Qg[(long)r * HEADDIM + kc]);
  }

  f32x8 zero = {0.f, 0.f, 0.f, 0.f, 0.f, 0.f, 0.f, 0.f};
  f32x8 O[8];
  float mrow[8], lrow[8];
#pragma unroll
  for (int i = 0; i < 8; i++) { O[i] = zero; mrow[i] = -1e30f; lrow[i] = 0.f; }

  for (int kt = 0; kt <= qt; kt++) {
    __syncthreads();  // previous tile fully consumed before overwrite
    const __bf16* Kg = Kc + ((long)kvh * SLEN + kt * 64) * HEADDIM;
#pragma unroll
    for (int i = 0; i < 8; i++) {
      int cid = tid + i * 128;
      int r = cid >> 4, kc = (cid & 15) << 3;
      async_ld_b128(&Ks[r * 128 + kc], &Kg[(long)r * HEADDIM + kc]);
    }
    const __bf16* Vg = Vt + (long)kvh * HEADDIM * SLEN + kt * 64;
#pragma unroll
    for (int i = 0; i < 8; i++) {
      int cid = tid + i * 128;
      int d = cid >> 3, nc = (cid & 7) << 3;
      async_ld_b128(&Vs[d * 64 + nc], &Vg[(long)d * SLEN + nc]);
    }
    wait_async0();
    __syncthreads();

    // S-tile: (16 q-rows per wave) x 64 keys
    f32x8 sacc[4];
#pragma unroll
    for (int nt = 0; nt < 4; nt++) sacc[nt] = zero;
#pragma unroll
    for (int ks = 0; ks < 128; ks += 32) {
      const __bf16* pa = &Qs[(w * 16 + l16) * 128 + ks + g * 16];
      bf16x8 alo = *(const bf16x8*)pa;
      bf16x8 ahi = *(const bf16x8*)(pa + 8);
      bf16x16 aq = __builtin_shufflevector(alo, ahi, 0, 1, 2, 3, 4, 5, 6, 7, 8,
                                           9, 10, 11, 12, 13, 14, 15);
#pragma unroll
      for (int nt = 0; nt < 4; nt++) {
        const __bf16* pb = &Ks[(nt * 16 + l16) * 128 + ks + g * 16];
        bf16x8 blo = *(const bf16x8*)pb;
        bf16x8 bhi = *(const bf16x8*)(pb + 8);
        bf16x16 bk = __builtin_shufflevector(blo, bhi, 0, 1, 2, 3, 4, 5, 6, 7,
                                             8, 9, 10, 11, 12, 13, 14, 15);
        sacc[nt] = __builtin_amdgcn_wmma_f32_16x16x32_bf16(
            false, aq, false, bk, (short)0, sacc[nt], false, false);
      }
    }

    // online softmax (row stats replicated across the 16 lanes of each g-half)
    float alpha[8];
#pragma unroll
    for (int j = 0; j < 8; j++) {
      int row = qBase + w * 16 + g * 8 + j;
      float mx = -1e30f;
#pragma unroll
      for (int nt = 0; nt < 4; nt++) {
        int col = kt * 64 + nt * 16 + l16;
        float sv = sacc[nt][j];
        if (col > row) sv -= 128.f;   // reference causal mask
        sacc[nt][j] = sv;
        mx = fmaxf(mx, sv);
      }
      for (int o = 1; o < 16; o <<= 1) mx = fmaxf(mx, __shfl_xor(mx, o, 32));
      float mnew = fmaxf(mrow[j], mx);
      alpha[j] = __expf(mrow[j] - mnew);
      mrow[j] = mnew;
      float rs = 0.f;
#pragma unroll
      for (int nt = 0; nt < 4; nt++) {
        float p = __expf(sacc[nt][j] - mnew);
        rs += p;
        Ps[(w * 16 + g * 8 + j) * 64 + nt * 16 + l16] = (__bf16)p;
      }
      for (int o = 1; o < 16; o <<= 1) rs += __shfl_xor(rs, o, 32);
      lrow[j] = lrow[j] * alpha[j] + rs;
    }
#pragma unroll
    for (int dt = 0; dt < 8; dt++)
#pragma unroll
      for (int j = 0; j < 8; j++) O[dt][j] = O[dt][j] * alpha[j];

    asm volatile("s_wait_dscnt 0x0" ::: "memory");  // P stores -> P fragment loads

    // O += P @ V   (Vs is d-major -> ready-made B^T tile)
#pragma unroll
    for (int ks = 0; ks < 64; ks += 32) {
      const __bf16* pa = &Ps[(w * 16 + l16) * 64 + ks + g * 16];
      bf16x8 alo = *(const bf16x8*)pa;
      bf16x8 ahi = *(const bf16x8*)(pa + 8);
      bf16x16 ap = __builtin_shufflevector(alo, ahi, 0, 1, 2, 3, 4, 5, 6, 7, 8,
                                           9, 10, 11, 12, 13, 14, 15);
#pragma unroll
      for (int dt = 0; dt < 8; dt++) {
        const __bf16* pv = &Vs[(dt * 16 + l16) * 64 + ks + g * 16];
        bf16x8 blo = *(const bf16x8*)pv;
        bf16x8 bhi = *(const bf16x8*)(pv + 8);
        bf16x16 bv = __builtin_shufflevector(blo, bhi, 0, 1, 2, 3, 4, 5, 6, 7,
                                             8, 9, 10, 11, 12, 13, 14, 15);
        O[dt] = __builtin_amdgcn_wmma_f32_16x16x32_bf16(
            false, ap, false, bv, (short)0, O[dt], false, false);
      }
    }
  }

#pragma unroll
  for (int dt = 0; dt < 8; dt++)
#pragma unroll
    for (int j = 0; j < 8; j++) {
      int row = qBase + w * 16 + g * 8 + j;
      int col = head * HEADDIM + dt * 16 + l16;
      Obf[(long)row * QDIM + col] = (__bf16)(O[dt][j] / lrow[j]);
    }
}

// ---------------------------------------------------------------------------
// Final RMSNorm of last token -> f32 vector
// ---------------------------------------------------------------------------
__global__ __launch_bounds__(256) void final_rms_kernel(
    const float* __restrict__ hidden, const float* __restrict__ w,
    float* __restrict__ out) {
  const float* row = hidden + (long)(SLEN - 1) * HDIM;
  float ss = 0.f;
  for (int h = threadIdx.x; h < HDIM; h += 256) { float v = row[h]; ss += v * v; }
  __shared__ float red[8];
  for (int o = 16; o; o >>= 1) ss += __shfl_xor(ss, o, 32);
  if ((threadIdx.x & 31) == 0) red[threadIdx.x >> 5] = ss;
  __syncthreads();
  if (threadIdx.x == 0) {
    float t = 0.f;
    for (int i = 0; i < 8; i++) t += red[i];
    red[0] = rsqrtf(t / (float)HDIM + EPSV);
  }
  __syncthreads();
  float inv = red[0];
  for (int h = threadIdx.x; h < HDIM; h += 256) out[h] = row[h] * inv * w[h];
}

// ---------------------------------------------------------------------------
// logits GEMV + block-level argmax partials; W is (H, V) row major
// ---------------------------------------------------------------------------
__global__ __launch_bounds__(256) void logits_argmax_part_kernel(
    const float* __restrict__ h, const float* __restrict__ W,
    float* __restrict__ pv, int* __restrict__ pi) {
  int v = blockIdx.x * 256 + threadIdx.x;
  float dot = -3e30f;
  if (v < VOCAB) {
    dot = 0.f;
    for (int k = 0; k < HDIM; k++) dot += h[k] * W[(long)k * VOCAB + v];
  }
  __shared__ float sv[256];
  __shared__ int si[256];
  sv[threadIdx.x] = dot;
  si[threadIdx.x] = v;
  __syncthreads();
  for (int st = 128; st; st >>= 1) {
    if (threadIdx.x < st) {
      float a = sv[threadIdx.x], b = sv[threadIdx.x + st];
      int ia = si[threadIdx.x], ib = si[threadIdx.x + st];
      if (b > a || (b == a && ib < ia)) { sv[threadIdx.x] = b; si[threadIdx.x] = ib; }
    }
    __syncthreads();
  }
  if (threadIdx.x == 0) { pv[blockIdx.x] = sv[0]; pi[blockIdx.x] = si[0]; }
}

__global__ __launch_bounds__(128) void argmax_final_kernel(
    const float* __restrict__ pv, const int* __restrict__ pi, int n,
    int* __restrict__ out) {
  float best = -3e30f;
  int bi = 0x7fffffff;
  for (int i = threadIdx.x; i < n; i += 128) {
    float v = pv[i];
    if (v > best || (v == best && pi[i] < bi)) { best = v; bi = pi[i]; }
  }
  __shared__ float sv[128];
  __shared__ int si[128];
  sv[threadIdx.x] = best;
  si[threadIdx.x] = bi;
  __syncthreads();
  for (int st = 64; st; st >>= 1) {
    if (threadIdx.x < st) {
      float a = sv[threadIdx.x], b = sv[threadIdx.x + st];
      int ia = si[threadIdx.x], ib = si[threadIdx.x + st];
      if (b > a || (b == a && ib < ia)) { sv[threadIdx.x] = b; si[threadIdx.x] = ib; }
    }
    __syncthreads();
  }
  if (threadIdx.x == 0) out[0] = si[0];
}

// ---------------------------------------------------------------------------
extern "C" void kernel_launch(void* const* d_in, const int* in_sizes, int n_in,
                              void* d_out, int out_size, void* d_ws, size_t ws_size,
                              hipStream_t stream) {
  (void)in_sizes; (void)n_in; (void)out_size; (void)ws_size;
  const int*     ids  = (const int*)d_in[2];
  const uint8_t* ed   = (const uint8_t*)d_in[5];
  const float*   zp   = (const float*)d_in[6];
  const float*   sc   = (const float*)d_in[7];
  const float*   cosb = (const float*)d_in[8];
  const float*   sinb = (const float*)d_in[9];
  const float*   ln1  = (const float*)d_in[10];
  const float*   qw   = (const float*)d_in[11];
  const float*   kw   = (const float*)d_in[12];
  const float*   vw   = (const float*)d_in[13];
  const float*   qnw  = (const float*)d_in[14];
  const float*   knw  = (const float*)d_in[15];
  const float*   ow   = (const float*)d_in[16];
  const float*   ln2  = (const float*)d_in[17];
  const float*   gw   = (const float*)d_in[18];
  const float*   uw   = (const float*)d_in[19];
  const float*   dw   = (const float*)d_in[20];
  const float*   fnw  = (const float*)d_in[21];
  const float*   lmw  = (const float*)d_in[22];

  char* ws = (char*)d_ws;
  size_t off = 0;
  auto alloc = [&](size_t bytes) -> void* {
    void* p = ws + off;
    off += (bytes + 255) & ~(size_t)255;
    return p;
  };
  float*  hidden = (float*)alloc((size_t)SLEN * HDIM * 4);
  __bf16* nbf    = (__bf16*)alloc((size_t)SLEN * HDIM * 2);
  float*  qf     = (float*)alloc((size_t)SLEN * QDIM * 4);
  float*  kf     = (float*)alloc((size_t)SLEN * KVDIM * 4);
  float*  vf     = (float*)alloc((size_t)SLEN * KVDIM * 4);
  __bf16* qbf    = (__bf16*)alloc((size_t)NHEADS * SLEN * HEADDIM * 2);
  __bf16* kbf    = (__bf16*)alloc((size_t)KVHEADS * SLEN * HEADDIM * 2);
  __bf16* vtb    = (__bf16*)alloc((size_t)KVHEADS * HEADDIM * SLEN * 2);
  __bf16* abf    = (__bf16*)alloc((size_t)SLEN * QDIM * 2);
  __bf16* wt     = (__bf16*)alloc((size_t)FFDIM * HDIM * 2);   // largest weight
  float*  gatef  = (float*)alloc((size_t)SLEN * FFDIM * 4);
  __bf16* actbf  = (__bf16*)alloc((size_t)SLEN * FFDIM * 2);
  float*  hvec   = (float*)alloc((size_t)HDIM * 4);
  const int nPart = (VOCAB + 255) / 256;
  float* pv = (float*)alloc((size_t)nPart * 4);
  int*   pi = (int*)alloc((size_t)nPart * 4);

  auto cvt = [&](const float* W, int K, int N) {
    long tot = (long)K * N;
    convert_wT_kernel<<<(unsigned)((tot + 255) / 256), 256, 0, stream>>>(W, wt, K, N);
  };
  auto gemm = [&](const __bf16* A, float* C, int M, int N, int K, int ep,
                  const float* gb, __bf16* ob) {
    dim3 grid(N / 128, M / 128);
    gemm_bf16_wmma<<<grid, 256, 0, stream>>>(A, wt, C, M, N, K, ep, gb, ob);
  };

  embed_kernel<<<(SLEN * HDIM) / 256, 256, 0, stream>>>(ids, ed, zp, sc, hidden);

  for (int i = 0; i < NLAYERS; i++) {
    rmsnorm_bf16_kernel<<<SLEN, 256, 0, stream>>>(hidden, ln1 + (long)i * HDIM, nbf);
    // Q, K, V projections
    cvt(qw + (long)i * HDIM * QDIM, HDIM, QDIM);
    gemm(nbf, qf, SLEN, QDIM, HDIM, 0, nullptr, nullptr);
    cvt(kw + (long)i * HDIM * KVDIM, HDIM, KVDIM);
    gemm(nbf, kf, SLEN, KVDIM, HDIM, 0, nullptr, nullptr);
    cvt(vw + (long)i * HDIM * KVDIM, HDIM, KVDIM);
    gemm(nbf, vf, SLEN, KVDIM, HDIM, 0, nullptr, nullptr);
    // head RMS + RoPE + layout packs
    qk_rope_kernel<<<SLEN * NHEADS, 128, 0, stream>>>(
        qf, qnw + (long)i * HEADDIM, cosb, sinb, qbf, NHEADS);
    qk_rope_kernel<<<SLEN * KVHEADS, 128, 0, stream>>>(
        kf, knw + (long)i * HEADDIM, cosb, sinb, kbf, KVHEADS);
    vtrans_kernel<<<(KVHEADS * SLEN * HEADDIM) / 256, 256, 0, stream>>>(vf, vtb);
    // attention
    flash_attn_kernel<<<NHEADS * (SLEN / 64), 128, 0, stream>>>(qbf, kbf, vtb, abf);
    // output projection + residual
    cvt(ow + (long)i * QDIM * HDIM, QDIM, HDIM);
    gemm(abf, hidden, SLEN, HDIM, QDIM, 1, nullptr, nullptr);
    // MLP
    rmsnorm_bf16_kernel<<<SLEN, 256, 0, stream>>>(hidden, ln2 + (long)i * HDIM, nbf);
    cvt(gw + (long)i * HDIM * FFDIM, HDIM, FFDIM);
    gemm(nbf, gatef, SLEN, FFDIM, HDIM, 0, nullptr, nullptr);
    cvt(uw + (long)i * HDIM * FFDIM, HDIM, FFDIM);
    gemm(nbf, gatef, SLEN, FFDIM, HDIM, 2, gatef, actbf);  // silu(gate)*up -> bf16
    cvt(dw + (long)i * FFDIM * HDIM, FFDIM, HDIM);
    gemm(actbf, hidden, SLEN, HDIM, FFDIM, 1, nullptr, nullptr);
  }

  final_rms_kernel<<<1, 256, 0, stream>>>(hidden, fnw, hvec);
  logits_argmax_part_kernel<<<nPart, 256, 0, stream>>>(hvec, lmw, pv, pi);
  argmax_final_kernel<<<1, 128, 0, stream>>>(pv, pi, nPart, (int*)d_out);
}
